// PolyHashV8_87016037416985
// MI455X (gfx1250) — matrix operand order repaired and checked by reading, not compiled, not run on hardware
//
#include <hip/hip_runtime.h>

#define B_ 32
#define T_ 512
#define D_ 128
#define H_ 512
#define E_ 16
#define M_ (B_*T_)      // 16384 rows
#define KF 160          // padded feature K (156 -> 160 = 5 x 32)
#define BUCKETS_ 65536

#define USE_ASYNC 1     // gfx1250 async global->LDS staging

typedef __bf16 bf16_t;
typedef bf16_t v16bf __attribute__((ext_vector_type(16)));
typedef float  v8f   __attribute__((ext_vector_type(8)));

__device__ __constant__ unsigned P32c[32] = {
  2654435761u,2246822519u,3266489917u,2028178513u,1220703125u,1610612741u,805306457u,402653189u,
  3674653429u,2860486313u,1073676287u,2971215073u,1500450271u,3267000013u,2654435789u,4049292737u,
  2246822531u,3266489927u,2028178519u,1220703133u,1610612759u,805306463u,402653201u,3674653441u,
  2860486319u,1073676311u,2971215091u,1500450277u,3267000023u,2654435801u,4049292751u,2246822537u};
__device__ __constant__ int MOFF[12] = {1,2,3,4,5,6,7,8,12,16,24,32};

__device__ inline bf16_t f2bf(float f){
  unsigned u = __builtin_bit_cast(unsigned, f);
  unsigned r = (u + 0x7FFFu + ((u >> 16) & 1u)) >> 16;
  return __builtin_bit_cast(bf16_t, (unsigned short)r);
}
__device__ inline float bf2f(bf16_t b){
  unsigned u = ((unsigned)__builtin_bit_cast(unsigned short, b)) << 16;
  return __builtin_bit_cast(float, u);
}
__device__ inline float silu_f(float x){ return x / (1.f + __expf(-x)); }

// ---------------- weight convert (f32 -> bf16, optional K padding) ----------------
__global__ void k_cvt(const float* __restrict__ src, bf16_t* __restrict__ dst,
                      int N, int Ks, int Kd){
  int i = blockIdx.x*blockDim.x + threadIdx.x;
  if (i >= N*Kd) return;
  int n = i / Kd, k = i - n*Kd;
  dst[i] = (k < Ks) ? f2bf(src[(size_t)n*Ks + k]) : f2bf(0.f);
}

// ---------------- hash + scale-embed attention ----------------
__global__ void k_hash_attn(const int* __restrict__ chars, const float* __restrict__ byte_tb,
                            const float* __restrict__ tables, const float* __restrict__ cond_w,
                            const float* __restrict__ q_w, const float* __restrict__ k_w,
                            float* __restrict__ hf_raw){
  int idx = blockIdx.x*blockDim.x + threadIdx.x;
  if (idx >= M_) return;
  int b = idx >> 9, t = idx & (T_-1);
  const int* ch = chars + (size_t)b*T_;

  unsigned key[8];
  #pragma unroll
  for (int s = 0; s < 5; ++s){            // scales 0..4 use P32 prefix directly
    int w = 1 << s; int jm = w < t ? w : t;
    unsigned acc = 0;
    for (int j = 0; j < jm; ++j) acc ^= (unsigned)ch[t-1-j] * P32c[j];
    key[s] = acc & 0xFFFFu;
  }
  { int jm = 32 < t ? 32 : t; unsigned acc = 0;
    for (int j = 0; j < jm; ++j) acc ^= (unsigned)ch[t-1-j] * (P32c[j] ^ 3735928559u);
    key[5] = acc & 0xFFFFu; }
  { int jm = 64 < t ? 64 : t; unsigned acc = 0;
    for (int j = 0; j < jm; ++j) acc ^= (unsigned)ch[t-1-j] * (P32c[j & 7] ^ 3405691582u);
    key[6] = acc & 0xFFFFu; }
  { int jm = 128 < t ? 128 : t; unsigned acc = 0;
    for (int j = 0; j < jm; ++j) acc ^= (unsigned)ch[t-1-j] * (P32c[j & 7] ^ 2343432205u);
    key[7] = acc & 0xFFFFu; }

  float se[8][16];
  #pragma unroll
  for (int s = 0; s < 4; ++s){
    const float* e = tables + ((size_t)s*BUCKETS_ + key[s])*E_;
    #pragma unroll
    for (int i = 0; i < 16; ++i) se[s][i] = e[i];
  }
  unsigned ck = 0;
  #pragma unroll
  for (int j = 0; j < 8; ++j){
    float l = 0.f;
    #pragma unroll
    for (int c = 0; c < 64; ++c) l += se[c >> 4][c & 15] * cond_w[j*64 + c];
    if (l > 0.f) ck ^= (P32c[j] & 0xFFFFu);
  }
  #pragma unroll
  for (int s = 4; s < 8; ++s){
    unsigned kk = key[s] ^ ck;
    const float* e = tables + ((size_t)s*BUCKETS_ + kk)*E_;
    #pragma unroll
    for (int i = 0; i < 16; ++i) se[s][i] = e[i];
  }
  const float* be = byte_tb + (size_t)ch[t]*D_;
  float Q[16];
  #pragma unroll
  for (int e = 0; e < 16; ++e) Q[e] = 0.f;
  for (int d = 0; d < 128; ++d){
    float bd = be[d];
    #pragma unroll
    for (int e = 0; e < 16; ++e) Q[e] += bd * q_w[e*128 + d];
  }
  float qk[16];
  #pragma unroll
  for (int e2 = 0; e2 < 16; ++e2){
    float s = 0.f;
    #pragma unroll
    for (int e = 0; e < 16; ++e) s += Q[e] * k_w[e*16 + e2];
    qk[e2] = s;
  }
  float sc[8]; float mx = -1e30f;
  #pragma unroll
  for (int n = 0; n < 8; ++n){
    float s = 0.f;
    #pragma unroll
    for (int i = 0; i < 16; ++i) s += se[n][i] * qk[i];
    sc[n] = s * 0.25f; mx = sc[n] > mx ? sc[n] : mx;
  }
  float den = 0.f;
  #pragma unroll
  for (int n = 0; n < 8; ++n){ sc[n] = __expf(sc[n] - mx); den += sc[n]; }
  float inv = 1.f / den;
  float* o = hf_raw + (size_t)idx*E_;
  #pragma unroll
  for (int i = 0; i < 16; ++i){
    float s = 0.f;
    #pragma unroll
    for (int n = 0; n < 8; ++n) s += sc[n] * se[n][i];
    o[i] = s * inv;
  }
}

// ---------------- depthwise conv + feature assembly (bf16) ----------------
__global__ void k_feats(const int* __restrict__ chars, const float* __restrict__ byte_tb,
                        const float* __restrict__ hf_raw, const float* __restrict__ conv_w,
                        const float* __restrict__ conv_b, bf16_t* __restrict__ feats){
  int idx = blockIdx.x*blockDim.x + threadIdx.x;
  if (idx >= M_) return;
  int b = idx >> 9, t = idx & (T_-1);
  const int* ch = chars + (size_t)b*T_;
  bf16_t* f = feats + (size_t)idx*KF;
  const float* be = byte_tb + (size_t)ch[t]*D_;
  for (int d = 0; d < 128; ++d) f[d] = f2bf(be[d]);
  #pragma unroll
  for (int e = 0; e < 16; ++e){
    float acc = conv_b[e];
    #pragma unroll
    for (int k = 0; k < 8; ++k){
      int tt = t - 7 + k;
      if (tt >= 0) acc += hf_raw[((size_t)b*T_ + tt)*E_ + e] * conv_w[e*8 + k];
    }
    f[128 + e] = f2bf(silu_f(acc));
  }
  #pragma unroll
  for (int i = 0; i < 12; ++i){
    int k = MOFF[i];
    f[144 + i] = f2bf((t >= k && ch[t] == ch[t-k]) ? 1.f : 0.f);
  }
  f[156] = f[157] = f[158] = f[159] = f2bf(0.f);
}

// ---------------- A-panel staging: global -> LDS (async on CDNA5) ----------------
// Copies 32 rows x K bf16 into LDS with padded stride lds_ld = K+8 (bank spread).
__device__ inline void stage_panel(const bf16_t* __restrict__ A, int ld, int m0, int K,
                                   bf16_t* sA, int lds_ld){
  const int cpr = K >> 3;                 // 16B chunks per row
  const int total = 32 * cpr;
  for (int i = threadIdx.x; i < total; i += 256){
    int r = i / cpr, c = i - r*cpr;
    const bf16_t* g = A + (size_t)(m0 + r)*ld + c*8;
    bf16_t* l = sA + r*lds_ld + c*8;
#if USE_ASYNC
    unsigned loff = (unsigned)(size_t)l;  // low 32 bits of flat shared ptr = LDS offset
    asm volatile("global_load_async_to_lds_b128 %0, %1, off"
                 :: "v"(loff), "v"(g) : "memory");
#else
    ((float4*)l)[0] = ((const float4*)g)[0];
#endif
  }
#if USE_ASYNC
  asm volatile("s_wait_asynccnt 0x0" ::: "memory");
#endif
  __syncthreads();
}

// ---------------- WMMA fragment loaders (bf16, 16x16x32) ----------------
__device__ inline v16bf frag_a_lds(const bf16_t* sA, int lds_ld, int mt, int k0, int lane){
  int m = lane & 15, h = lane >> 4;
  const bf16_t* p = sA + (mt*16 + m)*lds_ld + k0 + h*8;
  v16bf r;
  ((float4*)&r)[0] = ((const float4*)p)[0];        // K rel 0..7
  ((float4*)&r)[1] = ((const float4*)(p + 16))[0]; // K rel 16..23
  return r;
}
__device__ inline v16bf frag_b(const bf16_t* __restrict__ W, int ld, int n0, int k0, int lane){
  int n = lane & 15, h = lane >> 4;
  const bf16_t* p = W + (size_t)(n0 + n)*ld + k0 + h*8;
  v16bf r;
  ((float4*)&r)[0] = ((const float4*)p)[0];
  ((float4*)&r)[1] = ((const float4*)(p + 16))[0];
  return r;
}
#define WMMA_BF16(a,b,c) __builtin_amdgcn_wmma_f32_16x16x32_bf16(false,(a),false,(b),(short)0,(c),false,false)

// ---------------- GEMM1: h = silu(feats @ in_w^T + in_b)  (M-tile 32) ----------------
__global__ __launch_bounds__(256) void k_gemm_in(const bf16_t* __restrict__ A,
    const bf16_t* __restrict__ W, const float* __restrict__ bias, bf16_t* __restrict__ Hb){
  __shared__ __align__(16) bf16_t sA[32*(KF+8)];
  int lane = threadIdx.x & 31, wv = threadIdx.x >> 5;
  int m0 = blockIdx.x * 32, n0 = wv * 64;
  stage_panel(A, KF, m0, KF, sA, KF+8);
  v8f acc[2][4];
  #pragma unroll
  for (int mt = 0; mt < 2; ++mt) for (int t = 0; t < 4; ++t)
    for (int i = 0; i < 8; ++i) acc[mt][t][i] = 0.f;
  #pragma unroll
  for (int kk = 0; kk < 5; ++kk){
    int k0 = kk * 32;
    v16bf a0 = frag_a_lds(sA, KF+8, 0, k0, lane);
    v16bf a1 = frag_a_lds(sA, KF+8, 1, k0, lane);
    #pragma unroll
    for (int t = 0; t < 4; ++t){
      v16bf bfr = frag_b(W, KF, n0 + t*16, k0, lane);
      acc[0][t] = WMMA_BF16(a0, bfr, acc[0][t]);
      acc[1][t] = WMMA_BF16(a1, bfr, acc[1][t]);
    }
  }
  int nl = lane & 15, hh = lane >> 4;
  #pragma unroll
  for (int mt = 0; mt < 2; ++mt)
    #pragma unroll
    for (int t = 0; t < 4; ++t)
      #pragma unroll
      for (int i = 0; i < 8; ++i){
        int col = n0 + t*16 + nl, row = m0 + mt*16 + hh*8 + i;
        Hb[(size_t)row*H_ + col] = f2bf(silu_f(acc[mt][t][i] + bias[col]));
      }
}

// ---------------- GEMM2 fused gate: g = silu(h@r1^T) * (h@r2^T)  (M-tile 32) ----------------
__global__ __launch_bounds__(256) void k_gemm_gate(const bf16_t* __restrict__ Hb,
    const bf16_t* __restrict__ R1, const bf16_t* __restrict__ R2, bf16_t* __restrict__ G){
  __shared__ __align__(16) bf16_t sA[32*(H_+8)];
  int lane = threadIdx.x & 31, wv = threadIdx.x >> 5;
  int m0 = blockIdx.x * 32, n0 = wv * 64;
  stage_panel(Hb, H_, m0, H_, sA, H_+8);
  v8f a1[2][4], a2[2][4];
  #pragma unroll
  for (int mt = 0; mt < 2; ++mt) for (int t = 0; t < 4; ++t)
    for (int i = 0; i < 8; ++i){ a1[mt][t][i] = 0.f; a2[mt][t][i] = 0.f; }
  #pragma unroll 2
  for (int kk = 0; kk < 16; ++kk){
    int k0 = kk * 32;
    v16bf av0 = frag_a_lds(sA, H_+8, 0, k0, lane);
    v16bf av1 = frag_a_lds(sA, H_+8, 1, k0, lane);
    #pragma unroll
    for (int t = 0; t < 4; ++t){
      v16bf b1 = frag_b(R1, H_, n0 + t*16, k0, lane);
      a1[0][t] = WMMA_BF16(av0, b1, a1[0][t]);
      a1[1][t] = WMMA_BF16(av1, b1, a1[1][t]);
      v16bf b2 = frag_b(R2, H_, n0 + t*16, k0, lane);
      a2[0][t] = WMMA_BF16(av0, b2, a2[0][t]);
      a2[1][t] = WMMA_BF16(av1, b2, a2[1][t]);
    }
  }
  int nl = lane & 15, hh = lane >> 4;
  #pragma unroll
  for (int mt = 0; mt < 2; ++mt)
    #pragma unroll
    for (int t = 0; t < 4; ++t)
      #pragma unroll
      for (int i = 0; i < 8; ++i){
        int col = n0 + t*16 + nl, row = m0 + mt*16 + hh*8 + i;
        G[(size_t)row*H_ + col] = f2bf(silu_f(a1[mt][t][i]) * a2[mt][t][i]);
      }
}

// ---------------- GEMM3 + residual + LayerNorm  (M-tile 32) ----------------
__global__ __launch_bounds__(256) void k_gemm_resln(const bf16_t* __restrict__ G,
    const bf16_t* __restrict__ RP, const bf16_t* __restrict__ Hb,
    const float* __restrict__ ln_g, const float* __restrict__ ln_b, bf16_t* __restrict__ Z){
  __shared__ __align__(16) bf16_t sA[32*(H_+8)];
  __shared__ float s_sum[32], s_sq[32];
  int lane = threadIdx.x & 31, wv = threadIdx.x >> 5;
  int m0 = blockIdx.x * 32, n0 = wv * 64;
  stage_panel(G, H_, m0, H_, sA, H_+8);
  v8f acc[2][4];
  #pragma unroll
  for (int mt = 0; mt < 2; ++mt) for (int t = 0; t < 4; ++t)
    for (int i = 0; i < 8; ++i) acc[mt][t][i] = 0.f;
  #pragma unroll 2
  for (int kk = 0; kk < 16; ++kk){
    int k0 = kk * 32;
    v16bf av0 = frag_a_lds(sA, H_+8, 0, k0, lane);
    v16bf av1 = frag_a_lds(sA, H_+8, 1, k0, lane);
    #pragma unroll
    for (int t = 0; t < 4; ++t){
      v16bf bfr = frag_b(RP, H_, n0 + t*16, k0, lane);
      acc[0][t] = WMMA_BF16(av0, bfr, acc[0][t]);
      acc[1][t] = WMMA_BF16(av1, bfr, acc[1][t]);
    }
  }
  int nl = lane & 15, hh = lane >> 4;
  if (threadIdx.x < 32){ s_sum[threadIdx.x] = 0.f; s_sq[threadIdx.x] = 0.f; }
  // residual z = hb + h (in place on acc)
  #pragma unroll
  for (int mt = 0; mt < 2; ++mt)
    #pragma unroll
    for (int t = 0; t < 4; ++t)
      #pragma unroll
      for (int i = 0; i < 8; ++i){
        int col = n0 + t*16 + nl, row = m0 + mt*16 + hh*8 + i;
        acc[mt][t][i] += bf2f(Hb[(size_t)row*H_ + col]);
      }
  __syncthreads();
  #pragma unroll
  for (int mt = 0; mt < 2; ++mt)
    #pragma unroll
    for (int i = 0; i < 8; ++i){
      float ps = 0.f, pq = 0.f;
      #pragma unroll
      for (int t = 0; t < 4; ++t){ float v = acc[mt][t][i]; ps += v; pq += v*v; }
      int r = mt*16 + hh*8 + i;
      atomicAdd(&s_sum[r], ps);
      atomicAdd(&s_sq[r], pq);
    }
  __syncthreads();
  #pragma unroll
  for (int mt = 0; mt < 2; ++mt)
    #pragma unroll
    for (int i = 0; i < 8; ++i){
      int r = mt*16 + hh*8 + i;
      float mean = s_sum[r] * (1.f/512.f);
      float var  = s_sq[r] * (1.f/512.f) - mean*mean;
      float inv  = rsqrtf(var + 1e-5f);
      #pragma unroll
      for (int t = 0; t < 4; ++t){
        int col = n0 + t*16 + nl, row = m0 + mt*16 + hh*8 + i;
        Z[(size_t)row*H_ + col] = f2bf((acc[mt][t][i] - mean)*inv*ln_g[col] + ln_b[col]);
      }
    }
}

// ---------------- GEMM4: logits = z_ln @ out_w^T + out_b (f32 out, M-tile 32) ----------------
__global__ __launch_bounds__(256) void k_gemm_out(const bf16_t* __restrict__ Z,
    const bf16_t* __restrict__ WO, const float* __restrict__ ob, float* __restrict__ out){
  __shared__ __align__(16) bf16_t sA[32*(H_+8)];
  int lane = threadIdx.x & 31, wv = threadIdx.x >> 5;
  int m0 = blockIdx.x * 32, n0 = blockIdx.y * 512 + wv * 64;
  stage_panel(Z, H_, m0, H_, sA, H_+8);
  v8f acc[2][4];
  #pragma unroll
  for (int mt = 0; mt < 2; ++mt) for (int t = 0; t < 4; ++t)
    for (int i = 0; i < 8; ++i) acc[mt][t][i] = 0.f;
  #pragma unroll 2
  for (int kk = 0; kk < 16; ++kk){
    int k0 = kk * 32;
    v16bf av0 = frag_a_lds(sA, H_+8, 0, k0, lane);
    v16bf av1 = frag_a_lds(sA, H_+8, 1, k0, lane);
    #pragma unroll
    for (int t = 0; t < 4; ++t){
      v16bf bfr = frag_b(WO, H_, n0 + t*16, k0, lane);
      acc[0][t] = WMMA_BF16(av0, bfr, acc[0][t]);
      acc[1][t] = WMMA_BF16(av1, bfr, acc[1][t]);
    }
  }
  int nl = lane & 15, hh = lane >> 4;
  #pragma unroll
  for (int mt = 0; mt < 2; ++mt)
    #pragma unroll
    for (int t = 0; t < 4; ++t)
      #pragma unroll
      for (int i = 0; i < 8; ++i){
        int col = n0 + t*16 + nl, row = m0 + mt*16 + hh*8 + i;
        out[(size_t)row*1024 + col] = acc[mt][t][i] + ob[col];
      }
}

extern "C" void kernel_launch(void* const* d_in, const int* in_sizes, int n_in,
                              void* d_out, int out_size, void* d_ws, size_t ws_size,
                              hipStream_t stream){
  const int*   chars   = (const int*)  d_in[0];
  const float* byte_tb = (const float*)d_in[1];
  const float* tables  = (const float*)d_in[2];
  const float* cond_w  = (const float*)d_in[3];
  const float* q_w     = (const float*)d_in[4];
  const float* k_w     = (const float*)d_in[5];
  const float* conv_w  = (const float*)d_in[6];
  const float* conv_b  = (const float*)d_in[7];
  const float* in_w    = (const float*)d_in[8];
  const float* in_b    = (const float*)d_in[9];
  const float* r_w1    = (const float*)d_in[10];
  const float* r_w2    = (const float*)d_in[11];
  const float* r_proj  = (const float*)d_in[12];
  const float* ln_g    = (const float*)d_in[13];
  const float* ln_b    = (const float*)d_in[14];
  const float* out_w   = (const float*)d_in[15];
  const float* out_b   = (const float*)d_in[16];
  float* out = (float*)d_out;

  char* ws = (char*)d_ws;
  float*  hf    = (float*)(ws);                    // 16384*16*4   = 1 MB
  bf16_t* feats = (bf16_t*)(ws + 0x100000);        // 16384*160*2  = 5 MB
  bf16_t* Win   = (bf16_t*)(ws + 0x600000);        // 512*160*2
  bf16_t* R1    = (bf16_t*)(ws + 0x640000);        // 512*512*2
  bf16_t* R2    = (bf16_t*)(ws + 0x6C0000);
  bf16_t* RP    = (bf16_t*)(ws + 0x740000);
  bf16_t* WO    = (bf16_t*)(ws + 0x7C0000);        // 1024*512*2
  bf16_t* Hb    = (bf16_t*)(ws + 0x8C0000);        // 16384*512*2 = 16 MB
  bf16_t* G     = (bf16_t*)(ws + 0x18C0000);       // 16 MB
  bf16_t* Z     = (bf16_t*)(ws + 0x28C0000);       // 16 MB  (total ~59.5 MB)

  auto cdiv = [](int a, int b){ return (a + b - 1) / b; };
  k_cvt<<<cdiv(512*160,256),256,0,stream>>>(in_w,  Win, 512, 156, 160);
  k_cvt<<<cdiv(512*512,256),256,0,stream>>>(r_w1,  R1,  512, 512, 512);
  k_cvt<<<cdiv(512*512,256),256,0,stream>>>(r_w2,  R2,  512, 512, 512);
  k_cvt<<<cdiv(512*512,256),256,0,stream>>>(r_proj,RP,  512, 512, 512);
  k_cvt<<<cdiv(1024*512,256),256,0,stream>>>(out_w,WO, 1024, 512, 512);

  k_hash_attn<<<cdiv(M_,128),128,0,stream>>>(chars, byte_tb, tables, cond_w, q_w, k_w, hf);
  k_feats<<<cdiv(M_,128),128,0,stream>>>(chars, byte_tb, hf, conv_w, conv_b, feats);

  k_gemm_in   <<<M_/32, 256, 0, stream>>>(feats, Win, in_b, Hb);
  k_gemm_gate <<<M_/32, 256, 0, stream>>>(Hb, R1, R2, G);
  k_gemm_resln<<<M_/32, 256, 0, stream>>>(G, RP, Hb, ln_g, ln_b, Z);
  k_gemm_out  <<<dim3(M_/32, 2), 256, 0, stream>>>(Z, WO, out_b, out);
}
